// MultichannelMultiheadAttention_74534862455226
// MI455X (gfx1250) — matrix operand order repaired and checked by reading, not compiled, not run on hardware
//
#include <hip/hip_runtime.h>

// Problem constants (from reference)
#define B_    4
#define C_    2
#define Fdim  512
#define Wdim  1024
#define NH    8
#define HD    64
#define ROT   32
#define NFREQ 16
#define BC    (B_*C_)

typedef __attribute__((ext_vector_type(16))) __bf16         v16bf;
typedef __attribute__((ext_vector_type(8)))  float          v8f;
typedef __attribute__((ext_vector_type(8)))  unsigned short v8us;
typedef __attribute__((ext_vector_type(4)))  int            v4i;

// --------------------------------------------------------------------------
// Async global->LDS copy (GLOBAL_LOAD_ASYNC_TO_LDS_B128, ASYNCcnt-tracked).
// Prototype (probe-verified): (int4 AS1*, int4 AS3*, imm int, imm int)
// --------------------------------------------------------------------------
#if defined(__has_builtin)
#  if __has_builtin(__builtin_amdgcn_global_load_async_to_lds_b128)
#    define HAVE_ASYNC_LDS 1
#  endif
#endif
#ifndef HAVE_ASYNC_LDS
#  define HAVE_ASYNC_LDS 0
#endif

typedef v4i __attribute__((address_space(1)))* gptr4;
typedef v4i __attribute__((address_space(3)))* lptr4;

__device__ __forceinline__ void async_cp16(const unsigned short* g, unsigned short* l) {
#if HAVE_ASYNC_LDS
  __builtin_amdgcn_global_load_async_to_lds_b128((gptr4)g, (lptr4)l, 0, 0);
#else
  *(v8us*)l = *(const v8us*)g;
#endif
}

// Wait until at most `n` async ops remain (async loads complete in order,
// so <=2 after issuing next tile's 2 copies means the current tile is done).
__device__ __forceinline__ void async_wait_le2() {
#if HAVE_ASYNC_LDS
  asm volatile("s_wait_asynccnt 0x2" ::: "memory");
#endif
}
__device__ __forceinline__ void async_wait0() {
#if HAVE_ASYNC_LDS
  asm volatile("s_wait_asynccnt 0x0" ::: "memory");
#endif
}

__device__ __forceinline__ unsigned short f2bf_u(float f) {
  unsigned u = __float_as_uint(f);
  return (unsigned short)((u + 0x7FFFu + ((u >> 16) & 1u)) >> 16);
}

__device__ __forceinline__ v16bf mk16(v8us lo, v8us hi) {
  union { struct { v8us lo, hi; } p; v16bf v; } u;
  u.p.lo = lo; u.p.hi = hi;
  return u.v;
}

__device__ __forceinline__ v8f vzero8() {
  v8f z;
#pragma unroll
  for (int i = 0; i < 8; i++) z[i] = 0.f;
  return z;
}

// ---------------------------------------------------------------------------
// fp32 -> bf16 conversion
// ---------------------------------------------------------------------------
__global__ __launch_bounds__(256)
void cvt_f32_bf16(const float* __restrict__ in, unsigned short* __restrict__ out, int n) {
  int i = blockIdx.x * blockDim.x + threadIdx.x;
  if (i < n) out[i] = f2bf_u(in[i]);
}

// ---------------------------------------------------------------------------
// Batched per-channel GEMM:  Out[bc][g][w] = sum_h Wt[bc%C][g][h] * X[bc][h][w]
// M=Fdim (g), N=Wdim (w), K=Fdim (h). bf16 inputs, fp32 output.
// Workgroup tile 128x128, 8 waves (2x4), each wave 64x32 = 4x2 WMMA frags.
// Double-buffered pipeline: A tile via async global->LDS issued one tile
// ahead; B tile's global loads issued one tile ahead into registers.
// ---------------------------------------------------------------------------
#define GT_M 128
#define GT_N 128
#define GT_K 32
#define NKI  (Fdim / GT_K)   // 16 k-iterations
#define LDS_AS 40   // A tile LDS k-stride (pad, 80B rows keep 16B alignment)
#define LDS_BS 40   // B tile (transposed, [n][k]) k-stride

__global__ __launch_bounds__(256)
void gemm_bf16(const unsigned short* __restrict__ Wt,  // (C, Fdim, Fdim)
               const unsigned short* __restrict__ X,   // (BC, Fdim, Wdim)
               float* __restrict__ Out)                // (BC, Fdim, Wdim)
{
  __shared__ unsigned short lA[2][GT_M * LDS_AS];
  __shared__ unsigned short lB[2][GT_N * LDS_BS];

  const int bc = blockIdx.z;
  const int c  = bc % C_;
  const int m0 = blockIdx.y * GT_M;
  const int n0 = blockIdx.x * GT_N;
  const unsigned short* Wp = Wt + (size_t)c  * Fdim * Fdim;
  const unsigned short* Xp = X  + (size_t)bc * Fdim * Wdim;
  float*                Op = Out + (size_t)bc * Fdim * Wdim;

  const int t    = threadIdx.x;
  const int wid  = t >> 5;
  const int lane = t & 31;
  const int l15  = lane & 15;
  const int hi   = lane >> 4;        // 0: lanes 0-15, 1: lanes 16-31
  const int wm   = (wid >> 2) * 64;  // wave M offset in tile
  const int wn   = (wid & 3) * 32;   // wave N offset in tile

  // Per-thread staging coordinates
  const int aRow0 = t >> 2,           aCol0 = (t & 3) * 8;          // A seg 0
  const int aRow1 = (t + 256) >> 2,   aCol1 = ((t + 256) & 3) * 8;  // A seg 1
  const int bKr0  = t >> 4,           bNc0  = (t & 15) * 8;         // B seg 0
  const int bKr1  = (t + 256) >> 4,   bNc1  = ((t + 256) & 15) * 8; // B seg 1

  auto issueA = [&](int k0, int buf) {
    async_cp16(&Wp[(size_t)(m0 + aRow0) * Fdim + k0 + aCol0], &lA[buf][aRow0 * LDS_AS + aCol0]);
    async_cp16(&Wp[(size_t)(m0 + aRow1) * Fdim + k0 + aCol1], &lA[buf][aRow1 * LDS_AS + aCol1]);
  };
  auto loadB = [&](int k0, v8us breg[2]) {
    breg[0] = *(const v8us*)&Xp[(size_t)(k0 + bKr0) * Wdim + n0 + bNc0];
    breg[1] = *(const v8us*)&Xp[(size_t)(k0 + bKr1) * Wdim + n0 + bNc1];
  };
  auto storeB = [&](const v8us breg[2], int buf) {
#pragma unroll
    for (int e = 0; e < 8; e++) lB[buf][(bNc0 + e) * LDS_BS + bKr0] = breg[0][e];
#pragma unroll
    for (int e = 0; e < 8; e++) lB[buf][(bNc1 + e) * LDS_BS + bKr1] = breg[1][e];
  };

  v8f acc[4][2];
#pragma unroll
  for (int i = 0; i < 4; i++)
#pragma unroll
    for (int j = 0; j < 2; j++) acc[i][j] = vzero8();

  // Pipeline prologue: stage tile 0
  v8us breg[2];
  issueA(0, 0);
  loadB(0, breg);

#pragma unroll 2
  for (int it = 0; it < NKI; ++it) {
    const int cur = it & 1;
    const int k0  = it * GT_K;
    storeB(breg, cur);                      // B(it) regs -> LDS
    if (it + 1 < NKI) {                     // issue tile it+1 under compute
      issueA(k0 + GT_K, cur ^ 1);
      loadB(k0 + GT_K, breg);
      if (it + 2 < NKI) {                   // global_prefetch_b8, 2 tiles ahead
        __builtin_prefetch(&Wp[(size_t)(m0 + aRow0) * Fdim + k0 + 2 * GT_K], 0, 3);
        __builtin_prefetch(&Xp[(size_t)(k0 + 2 * GT_K + bKr0) * Wdim + n0 + bNc0], 0, 3);
      }
      async_wait_le2();                     // A(it) complete (in-order)
    } else {
      async_wait0();
    }
    __syncthreads();

    // B fragments: lane = n column; elems 0-7: k=kb.., elems 8-15: k=16+kb..
    v16bf bf[2];
#pragma unroll
    for (int j = 0; j < 2; j++) {
      int n  = wn + j * 16 + l15;
      int kb = hi * 8;
      v8us lo = *(const v8us*)&lB[cur][n * LDS_BS + kb];
      v8us hv = *(const v8us*)&lB[cur][n * LDS_BS + kb + 16];
      bf[j] = mk16(lo, hv);
    }
    // A fragments: lane = m row; elems 0-7: k=kl.., elems 8-15: k=16+kl..
#pragma unroll
    for (int i = 0; i < 4; i++) {
      int row = wm + i * 16 + l15;
      int kl  = hi * 8;
      v8us lo = *(const v8us*)&lA[cur][row * LDS_AS + kl];
      v8us hv = *(const v8us*)&lA[cur][row * LDS_AS + 16 + kl];
      v16bf af = mk16(lo, hv);
#pragma unroll
      for (int j = 0; j < 2; j++)
        acc[i][j] = __builtin_amdgcn_wmma_f32_16x16x32_bf16(
            false, af, false, bf[j], (short)0, acc[i][j], false, false);
    }
    __syncthreads();   // protect buffer `cur` before it is re-staged at it+2
  }

  // C layout: VGPR r -> row r (lanes 0-15) / row 8+r (lanes 16-31), col = lane&15
#pragma unroll
  for (int i = 0; i < 4; i++)
#pragma unroll
    for (int j = 0; j < 2; j++) {
      int col = n0 + wn + j * 16 + l15;
#pragma unroll
      for (int r = 0; r < 8; r++) {
        int row = m0 + wm + i * 16 + r + hi * 8;
        Op[(size_t)row * Wdim + col] = acc[i][j][r];
      }
    }
}

// ---------------------------------------------------------------------------
// Rope + head split.  P (BC, Fdim, Wdim) fp32 ->
//   Q (BC*NH, Wdim, HD) bf16  (rope'd; q == k in the reference)
//   V (BC*NH, HD, Wdim) bf16  (transposed so P·V B-frags are contiguous)
// ---------------------------------------------------------------------------
__global__ __launch_bounds__(256)
void rope_split(const float* __restrict__ P, const float* __restrict__ freqs,
                unsigned short* __restrict__ Q, unsigned short* __restrict__ V)
{
  int gid = blockIdx.x * blockDim.x + threadIdx.x;  // 0 .. BC*NH*Wdim-1
  int w   = gid % Wdim;
  int bcn = gid / Wdim;
  int bc  = bcn / NH;
  int n   = bcn % NH;
  const float* p = P + ((size_t)bc * Fdim + n * HD) * Wdim + w;  // stride Wdim over d
  unsigned short* q = Q + ((size_t)bcn * Wdim + w) * HD;
  unsigned short* v = V + (size_t)bcn * HD * Wdim + w;
  float pos = (float)w;
#pragma unroll
  for (int i = 0; i < NFREQ; i++) {
    float ang = pos * freqs[i];
    float cs = __cosf(ang), sn = __sinf(ang);
    float x1 = p[(size_t)(2 * i) * Wdim];
    float x2 = p[(size_t)(2 * i + 1) * Wdim];
    q[2 * i]     = f2bf_u(x1 * cs - x2 * sn);
    q[2 * i + 1] = f2bf_u(x2 * cs + x1 * sn);
    v[(size_t)(2 * i) * Wdim]     = f2bf_u(x1);
    v[(size_t)(2 * i + 1) * Wdim] = f2bf_u(x2);
  }
#pragma unroll
  for (int d = ROT; d < HD; d++) {
    float x = p[(size_t)d * Wdim];
    unsigned short b = f2bf_u(x);
    q[d] = b;
    v[(size_t)d * Wdim] = b;
  }
}

// ---------------------------------------------------------------------------
// Flash attention per (b,c,n): A = softmax(Q·Kᵀ/√Fdim)·V, K == Q.
// 8 waves/WG, each wave owns 16 query rows; double-buffered 32-key K/V
// tiles staged via async global->LDS one tile ahead of compute.
// Output written as (bc, h=n*HD+d, w=q) bf16, ready as GEMM2's X operand.
// ---------------------------------------------------------------------------
__global__ __launch_bounds__(256)
void attn(const unsigned short* __restrict__ Q,   // (BCN, Wdim, HD)
          const unsigned short* __restrict__ V,   // (BCN, HD, Wdim)
          unsigned short* __restrict__ A)         // (BC, Fdim, Wdim)
{
  __shared__ unsigned short lK[2][32 * HD];     // [key][d]
  __shared__ unsigned short lV[2][HD * 32];     // [d][key]
  __shared__ unsigned short lP[8 * 16 * 32];    // per-wave P relayout buffer

  const int n   = blockIdx.y;
  const int bc  = blockIdx.z;
  const int bcn = bc * NH + n;
  const unsigned short* Qp = Q + (size_t)bcn * Wdim * HD;
  const unsigned short* Vp = V + (size_t)bcn * HD * Wdim;

  const int t = threadIdx.x, wid = t >> 5, lane = t & 31;
  const int l15 = lane & 15, hi = lane >> 4;
  const int q0 = blockIdx.x * 128 + wid * 16;

  const int kKr = t >> 3, kSeg = (t & 7) * 8;   // K staging coords
  const int vD  = t >> 2, vSeg = (t & 3) * 8;   // V staging coords
  auto issueKV = [&](int kt, int buf) {
    async_cp16(&Qp[(size_t)(kt + kKr) * HD + kSeg], &lK[buf][kKr * HD + kSeg]);
    async_cp16(&Vp[(size_t)vD * Wdim + kt + vSeg], &lV[buf][vD * 32 + vSeg]);
  };

  // Q fragments (A-matrix, 16 rows x 32 d each), straight from global
  v16bf qa[2];
  {
    int qr = q0 + l15;
    int kl = hi * 8;
#pragma unroll
    for (int ds = 0; ds < 2; ds++) {
      v8us lo = *(const v8us*)&Qp[(size_t)qr * HD + ds * 32 + kl];
      v8us hv = *(const v8us*)&Qp[(size_t)qr * HD + ds * 32 + 16 + kl];
      qa[ds] = mk16(lo, hv);
    }
  }

  v8f acc[4];
#pragma unroll
  for (int j = 0; j < 4; j++) acc[j] = vzero8();
  float mrow[8], lrow[8];
#pragma unroll
  for (int r = 0; r < 8; r++) { mrow[r] = -1e30f; lrow[r] = 0.f; }
  const float scale = 0.044194173824159216f;  // 1/sqrt(512)

  issueKV(0, 0);  // pipeline prologue

#pragma unroll 2
  for (int it = 0; it < Wdim / 32; ++it) {
    const int cur = it & 1;
    if (it + 1 < Wdim / 32) {
      issueKV((it + 1) * 32, cur ^ 1);  // stage next tile under compute
      async_wait_le2();                 // current tile's 2 copies done
    } else {
      async_wait0();
    }
    __syncthreads();

    // S = Q·Kᵀ, two 16-key column tiles
    v8f s[2];
#pragma unroll
    for (int j = 0; j < 2; j++) {
      s[j] = vzero8();
      int key = j * 16 + l15;
#pragma unroll
      for (int ds = 0; ds < 2; ds++) {
        int db = ds * 32 + hi * 8;
        v8us lo = *(const v8us*)&lK[cur][key * HD + db];
        v8us hv = *(const v8us*)&lK[cur][key * HD + db + 16];
        v16bf kb = mk16(lo, hv);
        s[j] = __builtin_amdgcn_wmma_f32_16x16x32_bf16(
            false, qa[ds], false, kb, (short)0, s[j], false, false);
      }
    }

    // Online softmax: rows live in 16-lane halves -> width-16 xor reductions
    float fscale[8];
#pragma unroll
    for (int r = 0; r < 8; r++) {
      float mx = fmaxf(s[0][r], s[1][r]) * scale;
#pragma unroll
      for (int m = 8; m; m >>= 1) mx = fmaxf(mx, __shfl_xor(mx, m, 16));
      float mnew = fmaxf(mrow[r], mx);
      float f  = __expf(mrow[r] - mnew);
      float p0 = __expf(s[0][r] * scale - mnew);
      float p1 = __expf(s[1][r] * scale - mnew);
      s[0][r] = p0; s[1][r] = p1;
      float ps = p0 + p1;
#pragma unroll
      for (int m = 8; m; m >>= 1) ps += __shfl_xor(ps, m, 16);
      lrow[r] = lrow[r] * f + ps;
      mrow[r] = mnew;
      fscale[r] = f;
    }
#pragma unroll
    for (int j = 0; j < 4; j++)
#pragma unroll
      for (int r = 0; r < 8; r++) acc[j][r] *= fscale[r];

    // P: C-layout -> bf16 in LDS -> reload in A-layout (per-wave region)
    unsigned short* pp = &lP[wid * 16 * 32];
#pragma unroll
    for (int j = 0; j < 2; j++)
#pragma unroll
      for (int r = 0; r < 8; r++)
        pp[(r + hi * 8) * 32 + j * 16 + l15] = f2bf_u(s[j][r]);
    asm volatile("s_wait_dscnt 0" ::: "memory");
    v16bf pa;
    {
      int kl = hi * 8;
      v8us lo = *(const v8us*)&pp[l15 * 32 + kl];
      v8us hv = *(const v8us*)&pp[l15 * 32 + 16 + kl];
      pa = mk16(lo, hv);
    }
    // acc += P · V
#pragma unroll
    for (int j = 0; j < 4; j++) {
      int d  = j * 16 + l15;
      int kb = hi * 8;
      v8us lo = *(const v8us*)&lV[cur][d * 32 + kb];
      v8us hv = *(const v8us*)&lV[cur][d * 32 + kb + 16];
      v16bf vb = mk16(lo, hv);
      acc[j] = __builtin_amdgcn_wmma_f32_16x16x32_bf16(
          false, pa, false, vb, (short)0, acc[j], false, false);
    }
    __syncthreads();   // protect buffer `cur` before it is re-staged at it+2
  }

  // Normalize and store A in (bc, h=n*HD+d, w=q) layout for GEMM2
  unsigned short* Ap = A + (size_t)bc * Fdim * Wdim;
#pragma unroll
  for (int j = 0; j < 4; j++) {
    int d = n * HD + j * 16 + l15;
#pragma unroll
    for (int r = 0; r < 8; r++) {
      int qr = q0 + r + hi * 8;
      Ap[(size_t)d * Wdim + qr] = f2bf_u(acc[j][r] / lrow[r]);
    }
  }
}

// ---------------------------------------------------------------------------
// Host launcher
// ---------------------------------------------------------------------------
extern "C" void kernel_launch(void* const* d_in, const int* in_sizes, int n_in,
                              void* d_out, int out_size, void* d_ws, size_t ws_size,
                              hipStream_t stream) {
  const float* x  = (const float*)d_in[0];
  const float* wq = (const float*)d_in[1];
  // reference uses wq for q, k AND v; wk (d_in[2]) and wv (d_in[3]) are dead
  const float* wo = (const float*)d_in[4];
  const float* fr = (const float*)d_in[5];
  float* out = (float*)d_out;

  char* ws = (char*)d_ws;
  unsigned short* xbf = (unsigned short*)(ws);                       // 8 MiB
  unsigned short* wqb = (unsigned short*)(ws + (8ull  << 20));       // 1 MiB
  unsigned short* wob = (unsigned short*)(ws + (9ull  << 20));       // 1 MiB
  float*          P   = (float*)         (ws + (10ull << 20));       // 16 MiB
  unsigned short* Qb  = (unsigned short*)(ws + (26ull << 20));       // 8 MiB
  unsigned short* Vb  = (unsigned short*)(ws + (34ull << 20));       // 8 MiB
  unsigned short* Ab  = (unsigned short*)(ws + (42ull << 20));       // 8 MiB

  const int nx = B_ * C_ * Fdim * Wdim;   // 4194304
  const int nw = C_ * Fdim * Fdim;        // 524288
  cvt_f32_bf16<<<(nx + 255) / 256, 256, 0, stream>>>(x,  xbf, nx);
  cvt_f32_bf16<<<(nw + 255) / 256, 256, 0, stream>>>(wq, wqb, nw);
  cvt_f32_bf16<<<(nw + 255) / 256, 256, 0, stream>>>(wo, wob, nw);

  dim3 gg(Wdim / GT_N, Fdim / GT_M, BC);
  gemm_bf16<<<gg, 256, 0, stream>>>(wqb, xbf, P);

  rope_split<<<(BC * NH * Wdim) / 256, 256, 0, stream>>>(P, fr, Qb, Vb);

  dim3 ga(Wdim / 128, NH, BC);
  attn<<<ga, 256, 0, stream>>>(Qb, Vb, Ab);

  gemm_bf16<<<gg, 256, 0, stream>>>(wob, Ab, out);
}
